// MultiHeadAttention_65335042507464
// MI455X (gfx1250) — matrix operand order repaired and checked by reading, not compiled, not run on hardware
//
#include <hip/hip_runtime.h>
#include <hip/hip_bf16.h>
#include <stdint.h>

typedef __bf16 bf16_t;
typedef __attribute__((ext_vector_type(16))) __bf16 v16bf;
typedef __attribute__((ext_vector_type(8)))  float  v8f;
typedef unsigned int u32x4 __attribute__((ext_vector_type(4)));
typedef int          i32x8 __attribute__((ext_vector_type(8)));
typedef int          i32x4 __attribute__((ext_vector_type(4)));

#if defined(__has_builtin)
#if __has_builtin(__builtin_amdgcn_tensor_load_to_lds) && __has_builtin(__builtin_amdgcn_s_wait_tensorcnt)
#define HAVE_TDM 1
#endif
#endif
#ifndef HAVE_TDM
#define HAVE_TDM 0
#endif

#define HDIM   16
#define DMODEL 1024
#define BB     4
#define NN     2048
#define MM     2048

__device__ __forceinline__ bf16_t f2bf(float f) { return (bf16_t)f; }

enum OutMode { OUT_BHND = 0, OUT_BHDN = 1, OUT_ROW_F32 = 2 };

// C[rows,1024] = A[rows,1024] (f32 or bf16) x W[1024,1024] (f32) + bias, stored per MODE.
// A tile staged via global_load_async_to_lds_b128; W tile staged via TDM tensor_load_to_lds.
template <typename AT, int MODE>
__launch_bounds__(256)
__global__ void gemm_wmma_bf16(const AT* __restrict__ A, const float* __restrict__ W,
                               const float* __restrict__ bias, void* __restrict__ Out,
                               int seqLen)
{
    constexpr bool A_F32 = (sizeof(AT) == 4);
    // f32 path: 128 rows x 36 floats (144B row stride, 16B aligned)
    // bf16 path: 128 rows x 48 bf16  (96B row stride, 16B aligned)
    __shared__ __align__(16) unsigned char ldsA_raw[128 * 144];
    __shared__ __align__(16) float ldsWf[32][64];   // [k][n] f32 tile (8KB)

    const int tid  = threadIdx.x;
    const int wave = tid >> 5;
    const int lane = tid & 31;
    const int row0 = blockIdx.x * 128;
    const int col0 = blockIdx.y * 64;

    const unsigned ldsA_base = (unsigned)(uintptr_t)&ldsA_raw[0];

    v8f acc[4] = {};

    for (int kk = 0; kk < DMODEL; kk += 32) {
#if HAVE_TDM
        if (wave == 0) {
            // Tensor DMA: 32x64 f32 tile of W starting at [kk][col0] -> ldsWf (row-major, no pad)
            unsigned long long ga = (unsigned long long)(uintptr_t)&W[(size_t)kk * 1024 + col0];
            unsigned lds_off = (unsigned)(uintptr_t)&ldsWf[0][0];
            u32x4 g0;
            g0[0] = 1u;                                          // count=1 (valid user D#)
            g0[1] = lds_off;                                     // lds_addr
            g0[2] = (unsigned)(ga & 0xffffffffu);                // global_addr[31:0]
            g0[3] = (unsigned)((ga >> 32) & 0x01ffffffu)         // global_addr[56:32]
                    | 0x80000000u;                               // type=2 ("image")
            i32x8 g1;
            g1[0] = (2 << 16);            // data_size = 4B
            g1[1] = (int)(1024u << 16);   // tensor_dim0[15:0]
            g1[2] = (int)(1024u << 16);   // tensor_dim0[31:16]=0 | tensor_dim1[15:0]=1024
            g1[3] = (int)(64u << 16);     // tensor_dim1[31:16]=0 | tile_dim0=64
            g1[4] = 32;                   // tile_dim1=32 | tile_dim2=0
            g1[5] = 1024;                 // tensor_dim0_stride[31:0]
            g1[6] = 0;                    // stride0[47:32]=0 | stride1[15:0]=0
            g1[7] = 0;
            i32x4 gz  = {0, 0, 0, 0};                 // groups 2/3 unused (2-D tensor)
            i32x8 gz8 = {0, 0, 0, 0, 0, 0, 0, 0};     // extra group (6-arg toolchain form)
            __builtin_amdgcn_tensor_load_to_lds(g0, g1, gz, gz, gz8, 0);
        }
#else
        #pragma unroll
        for (int i = 0; i < 8; ++i) {
            int e = tid + 256 * i;
            int k = e >> 6, c = e & 63;
            ldsWf[k][c] = W[(size_t)(kk + k) * 1024 + col0 + c];
        }
#endif
        // stage A tile with async global->LDS copies (16B per lane per op)
        if (A_F32) {
            #pragma unroll
            for (int i = 0; i < 4; ++i) {
                int ch = tid + 256 * i;            // 1024 chunks of 16B (128 rows x 128B)
                int r  = ch >> 3, c4 = (ch & 7) * 4;
                unsigned long long ga =
                    (unsigned long long)(uintptr_t)&A[(size_t)(row0 + r) * DMODEL + kk + c4];
                unsigned lo = ldsA_base + (unsigned)(r * 144 + c4 * 4);
                asm volatile("global_load_async_to_lds_b128 %0, %1, off"
                             :: "v"(lo), "v"(ga) : "memory");
            }
        } else {
            #pragma unroll
            for (int i = 0; i < 2; ++i) {
                int ch = tid + 256 * i;            // 512 chunks of 16B (128 rows x 64B)
                int r  = ch >> 2, c8 = (ch & 3) * 8;
                unsigned long long ga =
                    (unsigned long long)(uintptr_t)&A[(size_t)(row0 + r) * DMODEL + kk + c8];
                unsigned lo = ldsA_base + (unsigned)(r * 96 + c8 * 2);
                asm volatile("global_load_async_to_lds_b128 %0, %1, off"
                             :: "v"(lo), "v"(ga) : "memory");
            }
        }
        asm volatile("s_wait_asynccnt 0x0" ::: "memory");
#if HAVE_TDM
        if (wave == 0) __builtin_amdgcn_s_wait_tensorcnt(0);
#endif
        __syncthreads();

        // A fragment: 16x32 for this wave's rows (convert f32->bf16 at build time)
        v16bf a;
        {
            int r  = wave * 16 + (lane & 15);
            int kh = (lane >> 4) * 16;
            if (A_F32) {
                const float* pA = (const float*)(const void*)ldsA_raw + (size_t)r * 36;
                #pragma unroll
                for (int i = 0; i < 8; ++i) { a[i] = f2bf(pA[kh + i]); a[8 + i] = f2bf(pA[kh + 8 + i]); }
            } else {
                const bf16_t* pA = (const bf16_t*)(const void*)ldsA_raw + (size_t)r * 48;
                #pragma unroll
                for (int i = 0; i < 8; ++i) { a[i] = pA[kh + i]; a[8 + i] = pA[kh + 8 + i]; }
            }
        }
        #pragma unroll
        for (int j = 0; j < 4; ++j) {
            v16bf b;
            int c  = j * 16 + (lane & 15);
            int kb = (lane >> 4) * 16;
            #pragma unroll
            for (int i = 0; i < 16; ++i) b[i] = f2bf(ldsWf[kb + i][c]);   // both lane halves hit the same 16 banks
            acc[j] = __builtin_amdgcn_wmma_f32_16x16x32_bf16(false, a, false, b,
                                                             (short)0, acc[j], false, false);
        }
        __syncthreads();
    }

    // epilogue: bias + store per layout mode
    #pragma unroll
    for (int j = 0; j < 4; ++j) {
        int   n  = col0 + j * 16 + (lane & 15);
        float bv = bias[n];
        int   mbase = row0 + wave * 16 + ((lane >> 4) ? 8 : 0);
        #pragma unroll
        for (int r = 0; r < 8; ++r) {
            int   m   = mbase + r;
            float val = acc[j][r] + bv;
            if (MODE == OUT_ROW_F32) {
                ((float*)Out)[(size_t)m * 1024 + n] = val;
            } else {
                int b = m / seqLen, s = m % seqLen;
                int h = n >> 6,     d = n & 63;
                size_t off;
                if (MODE == OUT_BHND) off = ((size_t)(b * HDIM + h) * seqLen + s) * 64 + d;
                else                  off = ((size_t)(b * HDIM + h) * 64 + d) * (size_t)seqLen + s;
                ((bf16_t*)Out)[off] = f2bf(val);
            }
        }
    }
}

// Flash attention: one wave per 16-query tile per (b,h). Q:[B,H,N,64] K:[B,H,M,64] Vt:[B,H,64,M]
__launch_bounds__(256)
__global__ void flash_attn_wmma(const bf16_t* __restrict__ Q, const bf16_t* __restrict__ K,
                                const bf16_t* __restrict__ Vt,
                                const float* __restrict__ pad_mask,
                                const float* __restrict__ word_mask,
                                bf16_t* __restrict__ O)
{
    __shared__ bf16_t ldsP[8][16][40];   // per-wave 16x32 P tile (padded)

    const int tid  = threadIdx.x;
    const int wave = tid >> 5;
    const int lane = tid & 31;

    int tileId = blockIdx.x * 8 + wave;
    int qt = tileId % (NN / 16);
    int bh = tileId / (NN / 16);
    int h  = bh % HDIM;
    int b  = bh / HDIM;
    int q0 = qt * 16;

    const size_t headQ = (size_t)(b * HDIM + h) * NN * 64;
    const size_t headK = (size_t)(b * HDIM + h) * MM * 64;
    const size_t headV = (size_t)(b * HDIM + h) * 64 * MM;

    // Q A-fragments for dk 0..31 and 32..63 (held in registers for whole key loop)
    v16bf aq0, aq1;
    {
        const bf16_t* qp = Q + headQ + (size_t)(q0 + (lane & 15)) * 64;
        int kh = (lane >> 4) * 16;
        #pragma unroll
        for (int i = 0; i < 8; ++i) {
            aq0[i] = qp[kh + i];      aq0[8 + i] = qp[8 + kh + i];
            aq1[i] = qp[32 + kh + i]; aq1[8 + i] = qp[40 + kh + i];
        }
    }

    v8f acc[4] = {};
    float mrow[8], lrow[8];
    #pragma unroll
    for (int r = 0; r < 8; ++r) { mrow[r] = -3.0e38f; lrow[r] = 0.f; }

    const int myrow_lo = q0 + ((lane >> 4) ? 8 : 0);   // row of acc VGPR r is myrow_lo + r

    for (int n0 = 0; n0 < MM; n0 += 32) {
        if (n0 + 32 < MM) {   // prefetch next K/V chunk -> global_prefetch_b8
            __builtin_prefetch(K  + headK + (size_t)(n0 + 32 + lane) * 64, 0, 1);
            __builtin_prefetch(Vt + headV + (size_t)lane * MM + n0 + 32,   0, 1);
        }
        // scores for two 16-key tiles, K-reduction over DK=64 (two WMMAs each)
        v8f s[2];
        #pragma unroll
        for (int t = 0; t < 2; ++t) {
            v16bf bk0, bk1;
            const bf16_t* kp = K + headK + (size_t)(n0 + t * 16 + (lane & 15)) * 64;
            int kb = (lane >> 4) * 16;
            #pragma unroll
            for (int i = 0; i < 16; ++i) { bk0[i] = kp[kb + i]; bk1[i] = kp[32 + kb + i]; }
            v8f z = {};
            z = __builtin_amdgcn_wmma_f32_16x16x32_bf16(false, aq0, false, bk0, (short)0, z, false, false);
            z = __builtin_amdgcn_wmma_f32_16x16x32_bf16(false, aq1, false, bk1, (short)0, z, false, false);
            s[t] = z;
        }
        // scale 1/sqrt(64) and apply pad/word masks
        #pragma unroll
        for (int t = 0; t < 2; ++t) {
            int   key = n0 + t * 16 + (lane & 15);
            float pm  = (1.f - pad_mask[b * MM + key]) * 1e12f;
            #pragma unroll
            for (int r = 0; r < 8; ++r) {
                int   row = myrow_lo + r;
                float wm  = (1.f - word_mask[((size_t)b * NN + row) * MM + key]) * 1e12f;
                s[t][r] = s[t][r] * 0.125f - pm - wm;
            }
        }
        // online softmax (row stats reduced across each 16-lane half)
        float alpha[8];
        #pragma unroll
        for (int r = 0; r < 8; ++r) {
            float mx = fmaxf(s[0][r], s[1][r]);
            #pragma unroll
            for (int off = 8; off >= 1; off >>= 1) mx = fmaxf(mx, __shfl_xor(mx, off, 16));
            float mnew = fmaxf(mrow[r], mx);
            alpha[r] = __expf(mrow[r] - mnew);
            mrow[r]  = mnew;
            float p0 = __expf(s[0][r] - mnew);
            float p1 = __expf(s[1][r] - mnew);
            s[0][r] = p0; s[1][r] = p1;
            float rs = p0 + p1;
            #pragma unroll
            for (int off = 8; off >= 1; off >>= 1) rs += __shfl_xor(rs, off, 16);
            lrow[r] = lrow[r] * alpha[r] + rs;
        }
        #pragma unroll
        for (int j = 0; j < 4; ++j)
            #pragma unroll
            for (int r = 0; r < 8; ++r) acc[j][r] *= alpha[r];

        // C-layout -> A-layout for P via per-wave LDS tile (same-wave DS is in-order)
        #pragma unroll
        for (int t = 0; t < 2; ++t) {
            int kcol  = t * 16 + (lane & 15);
            int rbase = (lane >> 4) ? 8 : 0;
            #pragma unroll
            for (int r = 0; r < 8; ++r) ldsP[wave][rbase + r][kcol] = f2bf(s[t][r]);
        }
        v16bf ap;
        {
            int r  = lane & 15;
            int kh = (lane >> 4) * 16;
            #pragma unroll
            for (int i = 0; i < 8; ++i) { ap[i] = ldsP[wave][r][kh + i]; ap[8 + i] = ldsP[wave][r][kh + 8 + i]; }
        }
        // acc += P x V  (V transposed: contiguous per-lane 32B loads)
        #pragma unroll
        for (int j = 0; j < 4; ++j) {
            v16bf bv;
            const bf16_t* vp = Vt + headV + (size_t)(j * 16 + (lane & 15)) * MM + n0 + (lane >> 4) * 16;
            #pragma unroll
            for (int i = 0; i < 16; ++i) bv[i] = vp[i];
            acc[j] = __builtin_amdgcn_wmma_f32_16x16x32_bf16(false, ap, false, bv,
                                                             (short)0, acc[j], false, false);
        }
    }

    // O[b, row, h*64+dv] bf16 (A-matrix of the output projection)
    #pragma unroll
    for (int j = 0; j < 4; ++j) {
        int dv = j * 16 + (lane & 15);
        #pragma unroll
        for (int r = 0; r < 8; ++r) {
            int   row = myrow_lo + r;
            float val = acc[j][r] / lrow[r];
            O[((size_t)b * NN + row) * 1024 + h * 64 + dv] = f2bf(val);
        }
    }
}

extern "C" void kernel_launch(void* const* d_in, const int* in_sizes, int n_in,
                              void* d_out, int out_size, void* d_ws, size_t ws_size,
                              hipStream_t stream)
{
    const float* query     = (const float*)d_in[0];
    const float* key       = (const float*)d_in[1];
    const float* value     = (const float*)d_in[2];
    const float* pad_mask  = (const float*)d_in[3];
    const float* word_mask = (const float*)d_in[4];
    const float* Wq = (const float*)d_in[5];
    const float* bq = (const float*)d_in[6];
    const float* Wk = (const float*)d_in[7];
    const float* bk = (const float*)d_in[8];
    const float* Wv = (const float*)d_in[9];
    const float* bv = (const float*)d_in[10];
    const float* Wo = (const float*)d_in[11];
    const float* bo = (const float*)d_in[12];

    // workspace: 4 x (8192 x 1024) bf16 buffers = 64 MB
    bf16_t* qb  = (bf16_t*)d_ws;
    bf16_t* kb  = qb  + (size_t)BB * NN * 1024;
    bf16_t* vtb = kb  + (size_t)BB * MM * 1024;
    bf16_t* ob  = vtb + (size_t)BB * MM * 1024;

    dim3 block(256);
    dim3 gridP(64, 16);   // 8192/128 row-blocks x 1024/64 col-blocks

    gemm_wmma_bf16<float,  OUT_BHND   ><<<gridP, block, 0, stream>>>(query, Wq, bq, qb,  NN);
    gemm_wmma_bf16<float,  OUT_BHND   ><<<gridP, block, 0, stream>>>(key,   Wk, bk, kb,  MM);
    gemm_wmma_bf16<float,  OUT_BHDN   ><<<gridP, block, 0, stream>>>(value, Wv, bv, vtb, MM);

    flash_attn_wmma<<<dim3((BB * HDIM * (NN / 16)) / 8), block, 0, stream>>>(
        qb, kb, vtb, pad_mask, word_mask, ob);

    gemm_wmma_bf16<bf16_t, OUT_ROW_F32><<<gridP, block, 0, stream>>>(ob, Wo, bo, d_out, NN);
}
